// EnrichAttention_81415400063695
// MI455X (gfx1250) — matrix operand order repaired and checked by reading, not compiled, not run on hardware
//
#include <hip/hip_runtime.h>

typedef unsigned short u16;
typedef __attribute__((ext_vector_type(16))) __bf16 v16bf;
typedef __attribute__((ext_vector_type(8)))  float  v8f;
typedef __attribute__((ext_vector_type(4)))  unsigned int u32x4;
typedef __attribute__((ext_vector_type(8)))  int i32x8;
typedef __attribute__((ext_vector_type(4)))  int i32x4;

#define B_  32
#define L1_ 512
#define L2_ 512
#define H_  256
#define A_  256
#define H3_ 768
#define H2_ 512

// GRU LDS layout (bytes)
#define LDS_HP   0         // [32][768] f32   96 KB
#define LDS_HF   98304     // [32][256] f32   32 KB
#define LDS_HB   131072    // [32][256] bf16  16 KB
#define LDS_WRES 147456    // [768][64] bf16  96 KB (whh k-cols 0..63, TDM-resident)
#define LDS_TOTAL 245760

__device__ __forceinline__ u16 f2bf(float f) {
  unsigned u = __float_as_uint(f);
  u += 0x7FFFu + ((u >> 16) & 1u);   // round-to-nearest-even
  return (u16)(u >> 16);
}

union BfVec { v16bf v; u32x4 q[2]; };

// One 16x16 f32 tile of A[16,K] x Bt[16,K]^T via v_wmma_f32_16x16x32_bf16.
// A layout (ISA 7.12.2): lane r holds row (r&15); elems 0..7 -> K = k0+half*8+e,
// elems 8..15 -> K = k0+16+half*8+e. B: lane holds col (r&15); same K mapping
// as A. Both are two 16B contiguous loads per lane.
__device__ __forceinline__ v8f tile_mma_bf16(const u16* Arow0, int lda,
                                             const u16* Brow0, int ldb, int K) {
  const int lane = threadIdx.x & 31;
  const int r   = lane & 15;
  const int h8  = (lane >> 4) << 3;   // 0 or 8
  const u16* a = Arow0 + (long)r * lda;
  const u16* b = Brow0 + (long)r * ldb;
  v8f acc = {0.f,0.f,0.f,0.f,0.f,0.f,0.f,0.f};
  for (int k0 = 0; k0 < K; k0 += 32) {
    BfVec av, bv;
    av.q[0] = *(const u32x4*)(a + k0 + h8);
    av.q[1] = *(const u32x4*)(a + k0 + 16 + h8);
    bv.q[0] = *(const u32x4*)(b + k0 + h8);
    bv.q[1] = *(const u32x4*)(b + k0 + 16 + h8);
    acc = __builtin_amdgcn_wmma_f32_16x16x32_bf16(false, av.v, false, bv.v,
                                                  (short)0, acc, false, false);
  }
  return acc;
}

// ---------------- elementwise / pack kernels ----------------

__global__ void k_cast_bf16(const float* __restrict__ src, u16* __restrict__ dst, long n) {
  long i = (long)blockIdx.x * blockDim.x + threadIdx.x;
  if (i < n) dst[i] = f2bf(src[i]);
}

// dst[z][c][r] = src[z][r][c]  (cast to bf16)
__global__ void k_transpose_cast(const float* __restrict__ src, u16* __restrict__ dst,
                                 int R, int C, long s_in, long s_out) {
  long z = blockIdx.z;
  long idx = (long)blockIdx.x * blockDim.x + threadIdx.x;
  if (idx >= (long)R * C) return;
  int rr = (int)(idx / C), cc = (int)(idx % C);
  dst[z * s_out + (long)cc * R + rr] = f2bf(src[z * s_in + idx]);
}

// g[row][0:256] = bf16(x1[row][0:256]); g row stride 512 (concat with ctx)
__global__ void k_pack_g_x1(const float* __restrict__ x1, u16* __restrict__ g) {
  long idx = (long)blockIdx.x * blockDim.x + threadIdx.x;   // B*L1*H
  long row = idx >> 8;
  int  c   = (int)(idx & 255);
  g[row * H2_ + c] = f2bf(x1[idx]);
}

// ---------------- GEMM kernels (block=256 -> 8 waves, 32x64 tile region) ----------------

template<bool RELU>
__global__ void __launch_bounds__(256)
k_gemm_bf16out(const u16* __restrict__ A, long asz, int lda,
               const u16* __restrict__ Bt, long bsz, int ldb,
               u16* __restrict__ C, long csz, int ldc, int K) {
  int z    = blockIdx.z;
  int wave = threadIdx.x >> 5, lane = threadIdx.x & 31;
  int m0 = blockIdx.y * 32 + (wave >> 2) * 16;
  int n0 = blockIdx.x * 64 + (wave & 3) * 16;
  v8f acc = tile_mma_bf16(A + z * asz + (long)m0 * lda, lda,
                          Bt + z * bsz + (long)n0 * ldb, ldb, K);
  int col = n0 + (lane & 15);
  int mh  = m0 + ((lane >> 4) << 3);
  #pragma unroll
  for (int rr = 0; rr < 8; ++rr) {
    float v = acc[rr];
    if (RELU) v = fmaxf(v, 0.f);
    C[z * csz + (long)(mh + rr) * ldc + col] = f2bf(v);
  }
}

// S[b][i][j] = (a1d[b,i,:] . a2[b,j,:]) * W[i,j]   (f32 out)
__global__ void __launch_bounds__(256)
k_gemm_scores(const u16* __restrict__ A, const u16* __restrict__ Bt,
              const float* __restrict__ W, float* __restrict__ S) {
  int b    = blockIdx.z;
  int wave = threadIdx.x >> 5, lane = threadIdx.x & 31;
  int m0 = blockIdx.y * 32 + (wave >> 2) * 16;
  int n0 = blockIdx.x * 64 + (wave & 3) * 16;
  v8f acc = tile_mma_bf16(A + (long)b * L1_ * A_ + (long)m0 * A_, A_,
                          Bt + (long)b * L2_ * A_ + (long)n0 * A_, A_, A_);
  int col = n0 + (lane & 15);
  int mh  = m0 + ((lane >> 4) << 3);
  float* Sb = S + (long)b * L1_ * L2_;
  #pragma unroll
  for (int rr = 0; rr < 8; ++rr) {
    int row = mh + rr;
    Sb[(long)row * L2_ + col] = acc[rr] * W[row * L2_ + col];
  }
}

// xproj[m][n] = g[m,:] . wih[n,:] + bih[n]   (f32 out)
__global__ void __launch_bounds__(256)
k_gemm_xproj(const u16* __restrict__ g, const u16* __restrict__ wihb,
             const float* __restrict__ bih, float* __restrict__ xp) {
  int wave = threadIdx.x >> 5, lane = threadIdx.x & 31;
  int m0 = blockIdx.y * 32 + (wave >> 2) * 16;
  int n0 = blockIdx.x * 64 + (wave & 3) * 16;
  v8f acc = tile_mma_bf16(g + (long)m0 * H2_, H2_, wihb + (long)n0 * H2_, H2_, H2_);
  int col = n0 + (lane & 15);
  int mh  = m0 + ((lane >> 4) << 3);
  float bias = bih[col];
  #pragma unroll
  for (int rr = 0; rr < 8; ++rr)
    xp[(long)(mh + rr) * H3_ + col] = acc[rr] + bias;
}

// ---------------- softmax over axis i (per (b,j) column), online in f32 ----------------

__global__ void k_softmax_stats(const float* __restrict__ S,
                                float* __restrict__ cmax, float* __restrict__ crcp) {
  int j  = blockIdx.x * blockDim.x + threadIdx.x;  // 0 .. B*L2-1
  int b  = j >> 9, jj = j & 511;
  const float* col = S + (long)b * L1_ * L2_ + jj;
  float m = -3.4e38f, s = 0.f;
  for (int i = 0; i < L1_; ++i) {
    float v = col[(long)i * L2_];
    if (v > m) { s = s * __expf(m - v) + 1.f; m = v; }
    else       { s += __expf(v - m); }
  }
  cmax[j] = m;
  crcp[j] = 1.f / s;
}

__global__ void k_apply_softmax(const float* __restrict__ S,
                                const float* __restrict__ cmax,
                                const float* __restrict__ crcp,
                                u16* __restrict__ P) {
  long idx = (long)blockIdx.x * blockDim.x + threadIdx.x;  // B*L1*L2
  int  j   = (int)(idx & 511);
  long b   = idx >> 18;                                    // / (L1*L2)
  float v  = __expf(S[idx] - cmax[b * L2_ + j]) * crcp[b * L2_ + j];
  P[idx] = f2bf(v);
}

// ---------------- GRU scan: 1 block, 1024 threads = 32 waves ----------------
// Per step: hp[32,768] = h_bf16[32,256] @ whh^T (96 WMMA tiles, 3/wave), then
// fused r/z/n gates in f32. h lives in LDS. whh K-columns 0..63 are parked in
// LDS once at kernel start via the Tensor Data Mover (tensor_load_to_lds,
// TENSORcnt); the remaining K-chunks stream bf16 from L2.
__global__ void __launch_bounds__(1024)
k_gru(const float* __restrict__ xproj,   // [B, L1, 3H]
      const u16*  __restrict__ whhb,     // [3H, H] bf16
      const float* __restrict__ bhh,     // [3H]
      float* __restrict__ out) {         // [B, L1, H]
  extern __shared__ char smem[];
  float* hp   = (float*)(smem + LDS_HP);    // [32][768]
  float* hf   = (float*)(smem + LDS_HF);    // [32][256]
  u16*   hb   = (u16*)  (smem + LDS_HB);    // [32][256]
  u16*   wres = (u16*)  (smem + LDS_WRES);  // [768][64]  whh[:, 0:64]

  const int tid  = threadIdx.x;
  const int wave = tid >> 5, lane = tid & 31;
  const int r    = lane & 15;
  const int h8   = (lane >> 4) << 3;

  #pragma unroll
  for (int q = 0; q < 8; ++q) { int i = tid + q * 1024; hf[i] = 0.f; hb[i] = 0; }

  // --- TDM: DMA whh[0:768, 0:64] (bf16) into LDS, issued by wave 0 only ---
  if (tid < 32) {
    unsigned long long ga = (unsigned long long)(uintptr_t)whhb;
    unsigned lds_off = (unsigned)(uintptr_t)wres;   // LDS aperture: addr[31:0] = LDS offset
    u32x4 g0;
    g0[0] = 1u;                                        // count=1 (valid descriptor)
    g0[1] = lds_off;                                   // D#.lds_addr
    g0[2] = (unsigned)(ga & 0xFFFFFFFFu);              // D#.global_addr lo
    g0[3] = (unsigned)((ga >> 32) & 0x01FFFFFFu)       // D#.global_addr hi (bits 120:64)
            | 0x80000000u;                             // D#.type = 2 ("image")
    i32x8 g1;
    g1[0] = 1 << 16;               // data_size = 1 -> 2-byte elements
    g1[1] = (int)(256u << 16);     // tensor_dim0 = 256 (bits 79:48)
    g1[2] = (int)(768u << 16);     // tensor_dim1 = 768 (bits 111:80)
    g1[3] = (int)(64u << 16);      // tile_dim0   = 64  (bits 127:112)
    g1[4] = 768;                   // tile_dim1   = 768 (bits 143:128)
    g1[5] = 256;                   // tensor_dim0_stride = 256 (bits 207:160)
    g1[6] = 0; g1[7] = 0;
    i32x4 gz = {0, 0, 0, 0};       // 2-D tensor: groups 2/3 unused
#if defined(__clang_major__) && (__clang_major__ >= 23)
    i32x8 gz8 = {0, 0, 0, 0, 0, 0, 0, 0};
    __builtin_amdgcn_tensor_load_to_lds(g0, g1, gz, gz, gz8, 0);
#else
    __builtin_amdgcn_tensor_load_to_lds(g0, g1, gz, gz, 0);
#endif
    __builtin_amdgcn_s_wait_tensorcnt(0);
  }
  __syncthreads();

  for (int t = 0; t < L1_; ++t) {
    // hp = h @ whh^T : tiles (2 x 48), wave handles 3 consecutive tiles
    #pragma unroll
    for (int q = 0; q < 3; ++q) {
      int tix = wave * 3 + q;
      int m0 = (tix & 1) * 16;
      int n0 = (tix >> 1) * 16;
      const u16* a  = hb   + (m0 + r) * H_;
      const u16* bl = wres + (n0 + r) * 64;            // LDS-resident K 0..63
      const u16* bg = whhb + (long)(n0 + r) * H_;      // L2 stream K 64..255
      v8f acc = {0.f,0.f,0.f,0.f,0.f,0.f,0.f,0.f};
      #pragma unroll
      for (int k0 = 0; k0 < 64; k0 += 32) {
        BfVec av, bv;
        av.q[0] = *(const u32x4*)(a + k0 + h8);
        av.q[1] = *(const u32x4*)(a + k0 + 16 + h8);
        bv.q[0] = *(const u32x4*)(bl + k0 + h8);
        bv.q[1] = *(const u32x4*)(bl + k0 + 16 + h8);
        acc = __builtin_amdgcn_wmma_f32_16x16x32_bf16(false, av.v, false, bv.v,
                                                      (short)0, acc, false, false);
      }
      #pragma unroll
      for (int k0 = 64; k0 < H_; k0 += 32) {
        BfVec av, bv;
        av.q[0] = *(const u32x4*)(a + k0 + h8);
        av.q[1] = *(const u32x4*)(a + k0 + 16 + h8);
        bv.q[0] = *(const u32x4*)(bg + k0 + h8);
        bv.q[1] = *(const u32x4*)(bg + k0 + 16 + h8);
        acc = __builtin_amdgcn_wmma_f32_16x16x32_bf16(false, av.v, false, bv.v,
                                                      (short)0, acc, false, false);
      }
      int col = n0 + r;
      int mh  = m0 + h8;
      #pragma unroll
      for (int rr = 0; rr < 8; ++rr)
        hp[(mh + rr) * H3_ + col] = acc[rr];
    }
    __syncthreads();

    // gates: 32*256 elements, 8 per thread (PyTorch order r,z,n)
    #pragma unroll
    for (int q = 0; q < 8; ++q) {
      int idx = tid + q * 1024;
      int b = idx >> 8, c = idx & 255;
      const float* xp = xproj + ((long)b * L1_ + t) * H3_;
      if (q == 0 && t + 1 < L1_)
        __builtin_prefetch(xp + H3_ + c, 0, 0);        // warm next step's xproj row
      float xr = xp[c], xz = xp[H_ + c], xn = xp[2 * H_ + c];
      float hr = hp[b * H3_ + c]          + bhh[c];
      float hz = hp[b * H3_ + H_ + c]     + bhh[H_ + c];
      float hn = hp[b * H3_ + 2 * H_ + c] + bhh[2 * H_ + c];
      float rg = 1.f / (1.f + __expf(-(xr + hr)));
      float zg = 1.f / (1.f + __expf(-(xz + hz)));
      float ng = tanhf(xn + rg * hn);
      float hv = (1.f - zg) * ng + zg * hf[idx];
      out[((long)b * L1_ + t) * H_ + c] = hv;
      hf[idx] = hv;
      hb[idx] = f2bf(hv);
    }
    __syncthreads();
  }
}

// ---------------- host ----------------

extern "C" void kernel_launch(void* const* d_in, const int* in_sizes, int n_in,
                              void* d_out, int out_size, void* d_ws, size_t ws_size,
                              hipStream_t stream) {
  const float* x1  = (const float*)d_in[0];
  const float* x2  = (const float*)d_in[1];
  const float* w1  = (const float*)d_in[2];
  const float* w2  = (const float*)d_in[3];
  const float* D   = (const float*)d_in[4];
  const float* W   = (const float*)d_in[5];
  const float* wih = (const float*)d_in[6];
  const float* whh = (const float*)d_in[7];
  const float* bih = (const float*)d_in[8];
  const float* bhh = (const float*)d_in[9];
  float* out = (float*)d_out;
  char*  ws  = (char*)d_ws;

  // workspace layout (bytes); g aliases x1b/x2b, xproj aliases S/stats/P
  const size_t o_g    = 0;                       // 16 MiB  [x1b | x2b] then g
  const size_t o_w1b  = 16777216;
  const size_t o_w2b  = o_w1b + 131072;
  const size_t o_Dtb  = o_w2b + 131072;
  const size_t o_wihb = o_Dtb + 131072;
  const size_t o_whhb = o_wihb + 786432;
  const size_t o_a1b  = o_whhb + 393216;
  const size_t o_a2b  = o_a1b + 8388608;
  const size_t o_a1db = o_a2b + 8388608;
  const size_t o_x2tb = o_a1db + 8388608;
  const size_t o_S    = o_x2tb + 8388608;        // 33.5 MiB f32 scores
  const size_t o_cmax = o_S + 33554432;
  const size_t o_crcp = o_cmax + 65536;
  const size_t o_pb   = o_crcp + 65536;          // 16.8 MiB bf16 softmax probs
  const size_t o_xproj = o_S;                    // alias: written after S/P consumed

  u16* g    = (u16*)(ws + o_g);
  u16* x1b  = g;
  u16* x2b  = (u16*)(ws + o_g + 8388608);
  u16* w1b  = (u16*)(ws + o_w1b);
  u16* w2b  = (u16*)(ws + o_w2b);
  u16* Dtb  = (u16*)(ws + o_Dtb);
  u16* wihb = (u16*)(ws + o_wihb);
  u16* whhb = (u16*)(ws + o_whhb);
  u16* a1b  = (u16*)(ws + o_a1b);
  u16* a2b  = (u16*)(ws + o_a2b);
  u16* a1db = (u16*)(ws + o_a1db);
  u16* x2tb = (u16*)(ws + o_x2tb);
  float* S    = (float*)(ws + o_S);
  float* cmax = (float*)(ws + o_cmax);
  float* crcp = (float*)(ws + o_crcp);
  u16*   pb   = (u16*)(ws + o_pb);
  float* xproj = (float*)(ws + o_xproj);

  const long nBLH = (long)B_ * L1_ * H_;   // 4,194,304

  // 1) casts to bf16
  k_cast_bf16<<<(unsigned)(nBLH / 256), 256, 0, stream>>>(x1, x1b, nBLH);
  k_cast_bf16<<<(unsigned)(nBLH / 256), 256, 0, stream>>>(x2, x2b, nBLH);
  k_cast_bf16<<<256, 256, 0, stream>>>(w1, w1b, (long)A_ * H_);
  k_cast_bf16<<<256, 256, 0, stream>>>(w2, w2b, (long)A_ * H_);
  k_cast_bf16<<<1536, 256, 0, stream>>>(wih, wihb, (long)H3_ * H2_);
  k_cast_bf16<<<768, 256, 0, stream>>>(whh, whhb, (long)H3_ * H_);
  // 2) transposes: Dt[n][k] = D[k][n]; x2t[b][h][j] = x2[b][j][h]
  k_transpose_cast<<<dim3(256, 1, 1), 256, 0, stream>>>(D, Dtb, A_, A_, 0, 0);
  k_transpose_cast<<<dim3(512, 1, B_), 256, 0, stream>>>(x2, x2tb, L2_, H_,
                                                         (long)L2_ * H_, (long)H_ * L2_);
  // 3) a1 = relu(x1 @ w1^T), a2 = relu(x2 @ w2^T)   M=16384 N=256 K=256
  k_gemm_bf16out<true><<<dim3(4, 512, 1), 256, 0, stream>>>(
      x1b, 0, H_, w1b, 0, H_, a1b, 0, A_, H_);
  k_gemm_bf16out<true><<<dim3(4, 512, 1), 256, 0, stream>>>(
      x2b, 0, H_, w2b, 0, H_, a2b, 0, A_, H_);
  // 4) a1d = a1 @ D (via Dt)
  k_gemm_bf16out<false><<<dim3(4, 512, 1), 256, 0, stream>>>(
      a1b, 0, A_, Dtb, 0, A_, a1db, 0, A_, A_);
  // 5) S = (a1d . a2) * W   per batch: M=N=512, K=256
  k_gemm_scores<<<dim3(8, 16, B_), 256, 0, stream>>>(a1db, a2b, W, S);
  // 6-7) softmax over i per (b,j) column
  k_softmax_stats<<<64, 256, 0, stream>>>(S, cmax, crcp);
  k_apply_softmax<<<32768, 256, 0, stream>>>(S, cmax, crcp, pb);
  // 8) g[:, 0:256] = x1
  k_pack_g_x1<<<(unsigned)(nBLH / 256), 256, 0, stream>>>(x1, g);
  // 9) g[:, 256:512] = ctx = P @ x2 (via x2t)  per batch: M=512, N=256, K=512
  k_gemm_bf16out<false><<<dim3(4, 16, B_), 256, 0, stream>>>(
      pb, (long)L1_ * L2_, L2_, x2tb, (long)H_ * L2_, L2_,
      g + H_, (long)L1_ * H2_, H2_, L2_);
  // 10) xproj = g @ wih^T + bih   M=16384, N=768, K=512
  k_gemm_xproj<<<dim3(12, 512, 1), 256, 0, stream>>>(g, wihb, bih, xproj);
  // 11) sequential GRU scan (single workgroup, 240 KB dynamic LDS, TDM-staged whh)
  k_gru<<<1, 1024, LDS_TOTAL, stream>>>(xproj, whhb, bhh, out);
}
